// RechitsGCN_55551107007133
// MI455X (gfx1250) — matrix operand order repaired and checked by reading, not compiled, not run on hardware
//
#include <hip/hip_runtime.h>
#include <hip/hip_bf16.h>

typedef __attribute__((ext_vector_type(16))) _Float16 v16h;
typedef __attribute__((ext_vector_type(8)))  float    v8f;

#define NGRAPH 512      // B*C = 8*64
#define NN     256
#define FIN    18
#define FPAD   32
#define HID    64
#define LN_EPS 1e-3f

// ---- output tuple offsets (floats): out, sa, dense, attn, adj ----
#define OUT_OFF   ((size_t)0)
#define SA_OFF    ((size_t)NGRAPH*HID)                 // 32768
#define DENSE_OFF (SA_OFF + (size_t)NGRAPH*NN*HID)     // 8421376
#define ATTN_OFF  (DENSE_OFF + (size_t)NGRAPH*NN*HID)  // 16809984
#define ADJ_OFF   (ATTN_OFF + (size_t)NGRAPH*NN*NN)    // 50364416

// ---- dynamic LDS arena byte offsets ----
#define OFF_BIG   0u        // 131072: adjK f16[256][256] | later S f32[64][256] (64KB) + sSA(16KB) + sH2(16KB)
#define OFF_Q     131072u   // f16[256][64]: f_het, then q
#define OFF_V     163840u   // f16[256][64]: G=f_het*norm3, then v
#define OFF_KT    196608u   // f16[64][256]: k transposed
#define OFF_OG    229376u   // f16[256][64]: out_gcn, then P f16[64][256]
#define OFF_XH    262144u   // f16[256][32] x padded; later W1h(8KB)+W2h(8KB)
#define OFF_W     278528u   // f16 staging 8KB (theta/W_t padded, Q/K/V)
#define OFF_SAH   286720u   // f16[64][64] normalized sa
#define OFF_H     294912u   // f16[64][64] h1
#define OFF_COORD 303104u   // f32[512]
#define OFF_MASK  305152u   // f32[256]
#define OFF_N3    306176u   // f32[256]
#define OFF_RED   307200u   // f32[256]
#define OFF_PAR   308224u   // f32[7*64] biases + LN params
#define OFF_OUTA  310272u   // f32[64]
#define OFF_MISC  310528u   // f32[16]
#define SMEM_BYTES 310592u  // ~303 KB of the 320 KB WGP LDS

// ---------------- WMMA helpers (gfx1250, wave32) ----------------
__device__ __forceinline__ v8f zero8() {
  v8f z;
#pragma unroll
  for (int i = 0; i < 8; ++i) z[i] = 0.f;
  return z;
}

__device__ __forceinline__ v8f wmma16(v16h a, v16h b, v8f c) {
  // D = A(16x32 f16) * B(32x16 f16) + C(16x16 f32)
  return __builtin_amdgcn_wmma_f32_16x16x32_f16(false, a, false, b, (short)0, c,
                                                false, false);
}

// A tile (16x32 f16) from row-major LDS; base = tile origin, ld in elements.
// Layout (ISA 7.12.2): lanes 0-15 -> M=0..15 (K=0..7,16..23), lanes 16-31 same M (K=8..15,24..31)
__device__ __forceinline__ v16h ldsA(const _Float16* base, int ld) {
  const int lane = threadIdx.x & 31;
  const _Float16* r0 = base + (lane & 15) * ld + ((lane >> 4) << 3);
  v16h a;
#pragma unroll
  for (int i = 0; i < 8; ++i) a[i] = r0[i];
#pragma unroll
  for (int i = 0; i < 8; ++i) a[8 + i] = r0[16 + i];
  return a;
}

// B tile (32x16 f16) from row-major LDS; lane holds col N=lane&15, K=0..7,16..23 (lanes 0-15) / K=8..15,24..31 (lanes 16-31).
__device__ __forceinline__ v16h ldsB(const _Float16* base, int ld) {
  const int lane = threadIdx.x & 31;
  const _Float16* r = base + (lane & 15) + ((lane >> 4) << 3) * ld;
  v16h b;
#pragma unroll
  for (int i = 0; i < 8; ++i) b[i] = r[i * ld];
#pragma unroll
  for (int i = 0; i < 8; ++i) b[8 + i] = r[(16 + i) * ld];
  return b;
}

// D tile (16x16 f32) stores: lane holds col n=lane&15, rows (lane>>4)*8 + r
__device__ __forceinline__ void stDf32(float* base, int ld, v8f d) {
  const int lane = threadIdx.x & 31;
  float* p = base + (((lane >> 4) << 3)) * ld + (lane & 15);
#pragma unroll
  for (int r = 0; r < 8; ++r) p[r * ld] = d[r];
}
__device__ __forceinline__ void stDf16(_Float16* base, int ld, v8f d) {
  const int lane = threadIdx.x & 31;
  _Float16* p = base + (((lane >> 4) << 3)) * ld + (lane & 15);
#pragma unroll
  for (int r = 0; r < 8; ++r) p[r * ld] = (_Float16)d[r];
}
// transposed f16 store: element (m,n) -> base[n*ldT + m]
__device__ __forceinline__ void stDf16T(_Float16* base, int ldT, v8f d) {
  const int lane = threadIdx.x & 31;
  _Float16* p = base + (lane & 15) * ldT + ((lane >> 4) << 3);
#pragma unroll
  for (int r = 0; r < 8; ++r) p[r] = (_Float16)d[r];
}

// ---------------- fused per-graph kernel ----------------
__global__ __launch_bounds__(256, 1) void rechits_gcn_fused(
    const float* __restrict__ xg, const float* __restrict__ maskg,
    const float* __restrict__ W_t, const float* __restrict__ b_t,
    const float* __restrict__ theta, const float* __restrict__ Qw,
    const float* __restrict__ Kw, const float* __restrict__ Vw,
    const float* __restrict__ W1, const float* __restrict__ b1,
    const float* __restrict__ W2, const float* __restrict__ b2,
    const float* __restrict__ l1g, const float* __restrict__ l1b,
    const float* __restrict__ l2g, const float* __restrict__ l2b,
    float* __restrict__ out) {
  extern __shared__ char smem[];
  _Float16* sAdjK = (_Float16*)(smem + OFF_BIG);          // [256][256]
  float*    sS    = (float*)(smem + OFF_BIG);             // [64][256]
  float*    sSA   = (float*)(smem + OFF_BIG + 65536);     // [64][64]
  float*    sH2   = (float*)(smem + OFF_BIG + 81920);     // [64][64]
  _Float16* sQ    = (_Float16*)(smem + OFF_Q);
  _Float16* sV    = (_Float16*)(smem + OFF_V);
  _Float16* sKT   = (_Float16*)(smem + OFF_KT);
  _Float16* sOG   = (_Float16*)(smem + OFF_OG);
  _Float16* sP    = (_Float16*)(smem + OFF_OG);           // [64][256]
  _Float16* sXh   = (_Float16*)(smem + OFF_XH);           // [256][32]
  _Float16* sW1h  = (_Float16*)(smem + OFF_XH);           // [64][64] (after phase C)
  _Float16* sW2h  = (_Float16*)(smem + OFF_XH + 8192);
  _Float16* sW    = (_Float16*)(smem + OFF_W);
  _Float16* sSAh  = (_Float16*)(smem + OFF_SAH);
  _Float16* sH    = (_Float16*)(smem + OFF_H);
  float* sCoord = (float*)(smem + OFF_COORD);
  float* sMask  = (float*)(smem + OFF_MASK);
  float* sN3    = (float*)(smem + OFF_N3);
  float* sRed   = (float*)(smem + OFF_RED);
  float* sPar   = (float*)(smem + OFF_PAR);  // b_t|b1|b2|l1g|l1b|l2g|l2b
  float* sOutA  = (float*)(smem + OFF_OUTA);
  float* sMisc  = (float*)(smem + OFF_MISC);

  const int g    = blockIdx.x;
  const int tid  = threadIdx.x;
  const int wave = tid >> 5;
  const int lane = tid & 31;
  const int lrow = (lane >> 4) << 3;  // D-tile row base for this lane
  const int lcol = lane & 15;         // D-tile col for this lane

  const float* xG = xg + (size_t)g * NN * FIN;
  const float* mG = maskg + (size_t)g * NN;
  float* adjG  = out + ADJ_OFF   + (size_t)g * NN * NN;
  float* attnG = out + ATTN_OFF  + (size_t)g * NN * NN;
  float* saG   = out + SA_OFF    + (size_t)g * NN * HID;
  float* dnG   = out + DENSE_OFF + (size_t)g * NN * HID;

  // ---- Phase A: stage x/mask/params, build adj, adj^3(f16), deg->norm^3 ----
  __builtin_prefetch(xG + tid * FIN, 0, 0);  // global_prefetch_b8
  {
    float m = mG[tid];
    sMask[tid] = m;
    float cx = xG[tid * FIN + 0], cy = xG[tid * FIN + 1];
    sCoord[2 * tid] = cx;
    sCoord[2 * tid + 1] = cy;
#pragma unroll
    for (int f = 0; f < FPAD; ++f)
      sXh[tid * FPAD + f] = (f < FIN) ? (_Float16)xG[tid * FIN + f] : (_Float16)0.f;
    if (tid < HID) {
      sPar[tid]           = b_t[tid];
      sPar[64 + tid]      = b1[tid];
      sPar[128 + tid]     = b2[tid];
      sPar[192 + tid]     = l1g[tid];
      sPar[256 + tid]     = l1b[tid];
      sPar[320 + tid]     = l2g[tid];
      sPar[384 + tid]     = l2b[tid];
      sOutA[tid]          = 0.f;
    }
  }
  __syncthreads();
  {
    float cx = sCoord[2 * tid], cy = sCoord[2 * tid + 1], mt = sMask[tid];
    float deg = 0.f;
    for (int m2 = 0; m2 < NN; ++m2) {
      float dx = cx - sCoord[2 * m2], dy = cy - sCoord[2 * m2 + 1];
      float dsq = fminf(fmaxf(dx * dx + dy * dy, 1e-12f), 1e12f);
      float a = __expf(-__fsqrt_rn(dsq)) * mt * sMask[m2];
      deg += a;
      adjG[tid * NN + m2] = a;
      sAdjK[tid * NN + m2] = (_Float16)(a * a * a);
    }
    float nrm = __frsqrt_rn(deg + 1e-6f);
    sN3[tid] = nrm * nrm * nrm;
    sRed[tid] = mt;
  }
  __syncthreads();
  for (int s2 = 128; s2 > 0; s2 >>= 1) {
    if (tid < s2) sRed[tid] += sRed[tid + s2];
    __syncthreads();
  }
  if (tid == 0) sMisc[0] = sRed[0];  // n_mask
  // stage padded theta (32x64 f16)
  for (int i = tid; i < FPAD * HID; i += 256) {
    int r = i / HID, c = i % HID;
    sW[i] = (r < FIN) ? (_Float16)theta[r * HID + c] : (_Float16)0.f;
  }
  __syncthreads();

  // ---- Phase B: f_het = x @ theta  (64 WMMA tiles, 8 per wave) ----
  for (int it = 0; it < 8; ++it) {
    int tIdx = wave * 8 + it, mt = tIdx >> 2, nt = tIdx & 3;
    v8f acc = zero8();
    acc = wmma16(ldsA(sXh + mt * 16 * FPAD, FPAD), ldsB(sW + nt * 16, HID), acc);
    stDf16(sQ + (mt * 16) * HID + nt * 16, HID, acc);  // sQ <- f_het
  }
  __syncthreads();

  // ---- Phase C: G = f_het*norm3; f_hom = relu((adjK@G)*norm3); gate mix ----
  for (int c = 0; c < HID; ++c)
    sV[tid * HID + c] = (_Float16)((float)sQ[tid * HID + c] * sN3[tid]);
  for (int i = tid; i < FPAD * HID; i += 256) {  // padded W_t
    int r = i / HID, c = i % HID;
    sW[i] = (r < FIN) ? (_Float16)W_t[r * HID + c] : (_Float16)0.f;
  }
  __syncthreads();
  for (int it = 0; it < 8; ++it) {
    int tIdx = wave * 8 + it, mt = tIdx >> 2, nt = tIdx & 3;
    v8f acc = zero8();
#pragma unroll
    for (int ks = 0; ks < 8; ++ks)
      acc = wmma16(ldsA(sAdjK + (mt * 16) * NN + ks * 32, NN),
                   ldsB(sV + (ks * 32) * HID + nt * 16, HID), acc);
    v8f ga = zero8();
    ga = wmma16(ldsA(sXh + mt * 16 * FPAD, FPAD), ldsB(sW + nt * 16, HID), ga);
#pragma unroll
    for (int r = 0; r < 8; ++r) {
      int row = mt * 16 + lrow + r, col = nt * 16 + lcol;
      float fhom = fmaxf(acc[r] * sN3[row], 0.f);
      float gate = 1.f / (1.f + __expf(-(ga[r] + sPar[col])));
      float fhet = (float)sQ[row * HID + col];
      sOG[row * HID + col] = (_Float16)(gate * fhom + (1.f - gate) * fhet);
    }
  }
  __syncthreads();

  // ---- Phase D: q,k^T,v = out_gcn @ {Q,K,V}; stage W1/W2 f16 ----
  for (int i = tid; i < HID * HID; i += 256) {
    sW[i]   = (_Float16)Qw[i];
    sW1h[i] = (_Float16)W1[i];
    sW2h[i] = (_Float16)W2[i];
  }
  __syncthreads();
  for (int it = 0; it < 8; ++it) {
    int tIdx = wave * 8 + it, mt = tIdx >> 2, nt = tIdx & 3;
    v8f acc = zero8();
    acc = wmma16(ldsA(sOG + mt * 16 * HID + 0, HID),  ldsB(sW + 0 * HID + nt * 16, HID), acc);
    acc = wmma16(ldsA(sOG + mt * 16 * HID + 32, HID), ldsB(sW + 32 * HID + nt * 16, HID), acc);
    stDf16(sQ + mt * 16 * HID + nt * 16, HID, acc);  // sQ <- q (f_het dead)
  }
  __syncthreads();
  for (int i = tid; i < HID * HID; i += 256) sW[i] = (_Float16)Kw[i];
  __syncthreads();
  for (int it = 0; it < 8; ++it) {
    int tIdx = wave * 8 + it, mt = tIdx >> 2, nt = tIdx & 3;
    v8f acc = zero8();
    acc = wmma16(ldsA(sOG + mt * 16 * HID + 0, HID),  ldsB(sW + 0 * HID + nt * 16, HID), acc);
    acc = wmma16(ldsA(sOG + mt * 16 * HID + 32, HID), ldsB(sW + 32 * HID + nt * 16, HID), acc);
    stDf16T(sKT + (nt * 16) * NN + mt * 16, NN, acc);  // sKT[d][n]
  }
  __syncthreads();
  for (int i = tid; i < HID * HID; i += 256) sW[i] = (_Float16)Vw[i];
  __syncthreads();
  for (int it = 0; it < 8; ++it) {
    int tIdx = wave * 8 + it, mt = tIdx >> 2, nt = tIdx & 3;
    v8f acc = zero8();
    acc = wmma16(ldsA(sOG + mt * 16 * HID + 0, HID),  ldsB(sW + 0 * HID + nt * 16, HID), acc);
    acc = wmma16(ldsA(sOG + mt * 16 * HID + 32, HID), ldsB(sW + 32 * HID + nt * 16, HID), acc);
    stDf16(sV + mt * 16 * HID + nt * 16, HID, acc);  // sV <- v (G dead)
  }
  __syncthreads();

  // ---- Phase E: attention + LN + FFN + LN + mean, in 4 row-quarters ----
  for (int q4 = 0; q4 < 4; ++q4) {
    const int r0 = q4 * 64;
    // scores S = (q@k^T)/8 + mask bias  -> sS f32 [64][256]
    for (int it = 0; it < 8; ++it) {
      int tIdx = wave * 8 + it, mt = tIdx >> 4, nt = tIdx & 15;
      v8f acc = zero8();
      acc = wmma16(ldsA(sQ + (r0 + mt * 16) * HID + 0, HID),  ldsB(sKT + 0 * NN + nt * 16, NN), acc);
      acc = wmma16(ldsA(sQ + (r0 + mt * 16) * HID + 32, HID), ldsB(sKT + 32 * NN + nt * 16, NN), acc);
      float bias = (1.f - sMask[nt * 16 + lcol]) * -1e9f;
#pragma unroll
      for (int r = 0; r < 8; ++r) acc[r] = acc[r] * 0.125f + bias;
      stDf32(sS + (mt * 16) * NN + nt * 16, NN, acc);
    }
    __syncthreads();
    // softmax rows (4 threads per row), write attn + P(f16)
    {
      int r = tid >> 2, seg = tid & 3;
      float* rowp = sS + r * NN + seg * 64;
      float mx = -3.0e38f;
      for (int i = 0; i < 64; ++i) mx = fmaxf(mx, rowp[i]);
      sRed[tid] = mx;
      __syncthreads();
      int b4 = (r << 2);
      float rmx = fmaxf(fmaxf(sRed[b4], sRed[b4 + 1]), fmaxf(sRed[b4 + 2], sRed[b4 + 3]));
      float s = 0.f;
      for (int i = 0; i < 64; ++i) s += __expf(rowp[i] - rmx);
      __syncthreads();
      sRed[tid] = s;
      __syncthreads();
      float inv = 1.f / (sRed[b4] + sRed[b4 + 1] + sRed[b4 + 2] + sRed[b4 + 3]);
      float* arow = attnG + (size_t)(r0 + r) * NN + seg * 64;
      _Float16* prow = sP + r * NN + seg * 64;
      for (int i = 0; i < 64; ++i) {
        float p = __expf(rowp[i] - rmx) * inv;
        arow[i] = p;
        prow[i] = (_Float16)p;
      }
    }
    __syncthreads();
    // O = P @ v -> sSA f32 [64][64]
    for (int it = 0; it < 2; ++it) {
      int tIdx = wave * 2 + it, mt = tIdx >> 2, nt = tIdx & 3;
      v8f acc = zero8();
#pragma unroll
      for (int ks = 0; ks < 8; ++ks)
        acc = wmma16(ldsA(sP + (mt * 16) * NN + ks * 32, NN),
                     ldsB(sV + (ks * 32) * HID + nt * 16, HID), acc);
      stDf32(sSA + (mt * 16) * HID + nt * 16, HID, acc);
    }
    __syncthreads();
    // LN1 * mask -> saG, sSA(f32, in place), sSAh(f16)
    {
      int r = tid >> 2, seg = tid & 3, b4 = (r << 2);
      float* rp = sSA + r * HID + seg * 16;
      float s = 0.f;
      for (int i = 0; i < 16; ++i) s += rp[i];
      sRed[tid] = s;
      __syncthreads();
      float mu = (sRed[b4] + sRed[b4 + 1] + sRed[b4 + 2] + sRed[b4 + 3]) * (1.f / HID);
      float v2 = 0.f;
      for (int i = 0; i < 16; ++i) { float d = rp[i] - mu; v2 += d * d; }
      __syncthreads();
      sRed[tid] = v2;
      __syncthreads();
      float var = (sRed[b4] + sRed[b4 + 1] + sRed[b4 + 2] + sRed[b4 + 3]) * (1.f / HID);
      float rs = __frsqrt_rn(var + LN_EPS);
      float mk = sMask[r0 + r];
      for (int i = 0; i < 16; ++i) {
        int c = seg * 16 + i;
        float y = ((rp[i] - mu) * rs * sPar[192 + c] + sPar[256 + c]) * mk;
        rp[i] = y;
        sSAh[r * HID + c] = (_Float16)y;
        saG[(size_t)(r0 + r) * HID + c] = y;
      }
    }
    __syncthreads();
    // h1 = relu(saN@W1 + b1) -> sH f16
    for (int it = 0; it < 2; ++it) {
      int tIdx = wave * 2 + it, mt = tIdx >> 2, nt = tIdx & 3;
      v8f acc = zero8();
      acc = wmma16(ldsA(sSAh + mt * 16 * HID + 0, HID),  ldsB(sW1h + 0 * HID + nt * 16, HID), acc);
      acc = wmma16(ldsA(sSAh + mt * 16 * HID + 32, HID), ldsB(sW1h + 32 * HID + nt * 16, HID), acc);
      float bb = sPar[64 + nt * 16 + lcol];
#pragma unroll
      for (int r = 0; r < 8; ++r) acc[r] = fmaxf(acc[r] + bb, 0.f);
      stDf16(sH + mt * 16 * HID + nt * 16, HID, acc);
    }
    __syncthreads();
    // h2 = relu(h1@W2 + b2) -> sH2 f32
    for (int it = 0; it < 2; ++it) {
      int tIdx = wave * 2 + it, mt = tIdx >> 2, nt = tIdx & 3;
      v8f acc = zero8();
      acc = wmma16(ldsA(sH + mt * 16 * HID + 0, HID),  ldsB(sW2h + 0 * HID + nt * 16, HID), acc);
      acc = wmma16(ldsA(sH + mt * 16 * HID + 32, HID), ldsB(sW2h + 32 * HID + nt * 16, HID), acc);
      float bb = sPar[128 + nt * 16 + lcol];
#pragma unroll
      for (int r = 0; r < 8; ++r) acc[r] = fmaxf(acc[r] + bb, 0.f);
      stDf32(sH2 + mt * 16 * HID + nt * 16, HID, acc);
    }
    __syncthreads();
    // LN2(h2 + saN) * mask -> dense global + sH2 in place
    {
      int r = tid >> 2, seg = tid & 3, b4 = (r << 2);
      float* hp = sH2 + r * HID + seg * 16;
      const float* sp = sSA + r * HID + seg * 16;
      float s = 0.f;
      for (int i = 0; i < 16; ++i) { float z = hp[i] + sp[i]; hp[i] = z; s += z; }
      sRed[tid] = s;
      __syncthreads();
      float mu = (sRed[b4] + sRed[b4 + 1] + sRed[b4 + 2] + sRed[b4 + 3]) * (1.f / HID);
      float v2 = 0.f;
      for (int i = 0; i < 16; ++i) { float d = hp[i] - mu; v2 += d * d; }
      __syncthreads();
      sRed[tid] = v2;
      __syncthreads();
      float var = (sRed[b4] + sRed[b4 + 1] + sRed[b4 + 2] + sRed[b4 + 3]) * (1.f / HID);
      float rs = __frsqrt_rn(var + LN_EPS);
      float mk = sMask[r0 + r];
      for (int i = 0; i < 16; ++i) {
        int c = seg * 16 + i;
        float y = ((hp[i] - mu) * rs * sPar[320 + c] + sPar[384 + c]) * mk;
        hp[i] = y;
        dnG[(size_t)(r0 + r) * HID + c] = y;
      }
    }
    __syncthreads();
    // accumulate column sums for masked mean
    if (tid < HID) {
      float s = 0.f;
      for (int r = 0; r < 64; ++r) s += sH2[r * HID + tid];
      sOutA[tid] += s;
    }
    __syncthreads();
  }

  if (tid < HID) {
    float nm = sMisc[0];
    out[(size_t)g * HID + tid] = (nm > 0.f) ? sOutA[tid] / nm : 0.f;
  }
}

extern "C" void kernel_launch(void* const* d_in, const int* in_sizes, int n_in,
                              void* d_out, int out_size, void* d_ws, size_t ws_size,
                              hipStream_t stream) {
  (void)in_sizes; (void)n_in; (void)d_ws; (void)ws_size; (void)out_size;
  const float* x     = (const float*)d_in[0];
  const float* mask  = (const float*)d_in[1];
  // d_in[2] = training (unused in inference path)
  const float* W_t   = (const float*)d_in[3];
  const float* b_t   = (const float*)d_in[4];
  const float* theta = (const float*)d_in[5];
  const float* Qw    = (const float*)d_in[6];
  const float* Kw    = (const float*)d_in[7];
  const float* Vw    = (const float*)d_in[8];
  const float* W1    = (const float*)d_in[9];
  const float* b1    = (const float*)d_in[10];
  const float* W2    = (const float*)d_in[11];
  const float* b2    = (const float*)d_in[12];
  const float* l1g   = (const float*)d_in[13];
  const float* l1b   = (const float*)d_in[14];
  const float* l2g   = (const float*)d_in[15];
  const float* l2b   = (const float*)d_in[16];
  float* outp = (float*)d_out;

  hipFuncSetAttribute(reinterpret_cast<const void*>(rechits_gcn_fused),
                      hipFuncAttributeMaxDynamicSharedMemorySize, (int)SMEM_BYTES);
  rechits_gcn_fused<<<dim3(NGRAPH), dim3(256), SMEM_BYTES, stream>>>(
      x, mask, W_t, b_t, theta, Qw, Kw, Vw, W1, b1, W2, b2,
      l1g, l1b, l2g, l2b, outp);
}
